// CPRPackedLinear_64330020159900
// MI455X (gfx1250) — matrix operand order repaired and checked by reading, not compiled, not run on hardware
//
#include <hip/hip_runtime.h>

typedef __attribute__((ext_vector_type(16))) _Float16 v16h;
typedef __attribute__((ext_vector_type(8)))  float    v8f;

#define IN_F    4096
#define OUT_F   11008
#define LDS_PAD 72      // halfs per n-row: 64 data + 8 pad (144B rows, 16B aligned)

static __device__ __forceinline__ unsigned int pkh(_Float16 a, _Float16 b) {
  union { _Float16 h[2]; unsigned int u; } u_;
  u_.h[0] = a; u_.h[1] = b;
  return u_.u;
}

// ---------------------------------------------------------------------------
// Kernel 0: permute-gather x via col_indices, convert f32->f16, and emit the
// WMMA A-operand (16x32 f16) register image, contiguous k order (k = 16*hi + elem).
// Storage: dword index e = ((sub*4 + t)*32 + lane)*8 + i   (sub = 32-wide K step)
// ---------------------------------------------------------------------------
__global__ void cpr_build_a_frags(const float* __restrict__ x,
                                  const int*   __restrict__ ci,
                                  unsigned int* __restrict__ frag) {
  int e    = blockIdx.x * blockDim.x + threadIdx.x;   // 0 .. 131071
  int i    = e & 7;
  int lane = (e >> 3) & 31;
  int q    = e >> 8;                                  // sub*4 + t
  int t    = q & 3;
  int sub  = q >> 2;                                  // 0..127
  int m    = t * 16 + (lane & 15);
  int hi   = (lane >> 4) & 1;
  int kb   = 2 * i + (hi ? 16 : 0);
  int k    = sub * 32 + kb;                           // permuted K position
  _Float16 h0 = (_Float16)x[m * IN_F + ci[k]];
  _Float16 h1 = (_Float16)x[m * IN_F + ci[k + 1]];
  frag[e] = pkh(h0, h1);
}

// ---------------------------------------------------------------------------
// Fused dequant + GEMM. Grid = 344 (N/32), block = 128 (4 waves).
// Wave w: N columns n0 + 16*(w&1), M tiles {2*(w>>1), 2*(w>>1)+1}.
// K loop: 64-wide steps; steps 0..15 = 6-bit region, 16..63 = 5-bit region.
// Raw packed dwords are register-prefetched TWO steps ahead (ping-pong rA/rB),
// unpack+LDS-store runs one step ahead, compute consumes current step.
// ---------------------------------------------------------------------------
__global__ __launch_bounds__(128)
void cpr_gemm(const unsigned int* __restrict__ frag,
              const int*   __restrict__ Wh,    // (768, 11008)  packed bytes in i32
              const int*   __restrict__ Wl,    // (1920, 11008) packed bytes in i32
              const float* __restrict__ sh,    // (8, 11008)
              const float* __restrict__ slo,   // (24, 11008)
              const float* __restrict__ bias,  // (11008)
              float*       __restrict__ out) { // (64, 11008)
  __shared__ __align__(16) _Float16 lds[2][32 * LDS_PAD];

  const int tid   = threadIdx.x;
  const int lane  = tid & 31;
  const int wave  = tid >> 5;        // 0..3
  const int nsub  = wave & 1;        // which 16-col group
  const int mhalf = wave >> 1;       // which pair of M tiles
  const int n0    = blockIdx.x * 32;
  const int c     = tid & 31;        // staging: column within N tile
  const int ty    = tid >> 5;        // staging: 0..3

  v8f acc[2] = {v8f{0.f}, v8f{0.f}};

  // ---- fetch raw packed dwords for step s into registers ----
  auto fetch = [&](int s, unsigned int* r, float& sc) {
    if (s < 16) {                                     // 6-bit: 4 groups-of-4
      sc = sh[(s >> 1) * OUT_F + n0 + c];
      #pragma unroll
      for (int j = 0; j < 4; ++j) {
        const int g = ty + 4 * j;                     // local 4-group, 0..15
        const int* p = Wh + (size_t)(3 * (16 * s + g)) * OUT_F + (n0 + c);
        r[3 * j + 0] = (unsigned int)p[0];
        r[3 * j + 1] = (unsigned int)p[OUT_F];
        r[3 * j + 2] = (unsigned int)p[2 * OUT_F];
      }
    } else {                                          // 5-bit: 2 groups-of-8
      const int sr = s - 16;
      sc = slo[(sr >> 1) * OUT_F + n0 + c];
      #pragma unroll
      for (int j = 0; j < 2; ++j) {
        const int g = ty + 4 * j;                     // local 8-group, 0..7
        const int* p = Wl + (size_t)(5 * (8 * sr + g)) * OUT_F + (n0 + c);
        #pragma unroll
        for (int q = 0; q < 5; ++q) r[5 * j + q] = (unsigned int)p[q * OUT_F];
      }
    }
  };

  // ---- unpack raw dwords, dequantize, store f16 tile [n][k] into LDS ----
  auto unpackStore = [&](int s, const unsigned int* r, float sc,
                         _Float16* __restrict__ buf) {
    if (s < 16) {
      #pragma unroll
      for (int j = 0; j < 4; ++j) {
        const int g = ty + 4 * j;
        const unsigned int b0 = r[3 * j + 0] & 255u;
        const unsigned int b1 = r[3 * j + 1] & 255u;
        const unsigned int b2 = r[3 * j + 2] & 255u;
        const int v0 = (int)( b0        & 63u);
        const int v1 = (int)(((b0 >> 6) &  3u) | ((b1 & 15u) << 2));
        const int v2 = (int)(((b1 >> 4) & 15u) | ((b2 &  3u) << 4));
        const int v3 = (int)( (b2 >> 2) & 63u);
        uint2 d;
        d.x = pkh((_Float16)((float)(v0 - 31) * sc), (_Float16)((float)(v1 - 31) * sc));
        d.y = pkh((_Float16)((float)(v2 - 31) * sc), (_Float16)((float)(v3 - 31) * sc));
        *reinterpret_cast<uint2*>(&buf[c * LDS_PAD + 4 * g]) = d;    // 8B aligned
      }
    } else {
      #pragma unroll
      for (int j = 0; j < 2; ++j) {
        const int g = ty + 4 * j;
        const unsigned int b0 = r[5 * j + 0] & 255u;
        const unsigned int b1 = r[5 * j + 1] & 255u;
        const unsigned int b2 = r[5 * j + 2] & 255u;
        const unsigned int b3 = r[5 * j + 3] & 255u;
        const unsigned int b4 = r[5 * j + 4] & 255u;
        const int v0 = (int)( b0        & 31u);
        const int v1 = (int)(((b0 >> 5) &  7u) | ((b1 &  3u) << 3));
        const int v2 = (int)( (b1 >> 2) & 31u);
        const int v3 = (int)(((b1 >> 7) &  1u) | ((b2 & 15u) << 1));
        const int v4 = (int)(((b2 >> 4) & 15u) | ((b3 &  1u) << 4));
        const int v5 = (int)( (b3 >> 1) & 31u);
        const int v6 = (int)(((b3 >> 6) &  3u) | ((b4 &  7u) << 2));
        const int v7 = (int)( (b4 >> 3) & 31u);
        uint4 d;
        d.x = pkh((_Float16)((float)(v0 - 15) * sc), (_Float16)((float)(v1 - 15) * sc));
        d.y = pkh((_Float16)((float)(v2 - 15) * sc), (_Float16)((float)(v3 - 15) * sc));
        d.z = pkh((_Float16)((float)(v4 - 15) * sc), (_Float16)((float)(v5 - 15) * sc));
        d.w = pkh((_Float16)((float)(v6 - 15) * sc), (_Float16)((float)(v7 - 15) * sc));
        *reinterpret_cast<uint4*>(&buf[c * LDS_PAD + 8 * g]) = d;    // 16B aligned
      }
    }
  };

  // ---- consume one 64-K step from LDS: 4 WMMAs per wave ----
  auto compute = [&](int s, const _Float16* __restrict__ buf) {
    const int nl  = nsub * 16 + (lane & 15);
    const int khi = (lane >> 4) ? 16 : 0;             // B: lanes 16-31 hold K 16-31
    #pragma unroll
    for (int sub = 0; sub < 2; ++sub) {
      const v16h b = *reinterpret_cast<const v16h*>(
          &buf[nl * LDS_PAD + sub * 32 + khi]);       // conflict-free, 16B aligned
      const int sg = s * 2 + sub;                     // 32-wide K substep
      #pragma unroll
      for (int t2 = 0; t2 < 2; ++t2) {
        const int t = mhalf * 2 + t2;
        const v16h a = *reinterpret_cast<const v16h*>(
            frag + (size_t)((sg * 4 + t) * 32 + lane) * 8);  // coalesced 32B/lane
        acc[t2] = __builtin_amdgcn_wmma_f32_16x16x32_f16(
            false, a, false, b, (short)0, acc[t2], false, false);
      }
    }
  };

  // ---- software pipeline: fetch 2 ahead, unpack 1 ahead, compute current ----
  unsigned int rA[12]; float scA;
  unsigned int rB[12]; float scB;
  fetch(0, rA, scA);
  fetch(1, rB, scB);
  unpackStore(0, rA, scA, lds[0]);
  __syncthreads();

  #pragma unroll 1
  for (int s = 0; s < 64; s += 2) {
    // even phase: rB holds raw(s+1)
    if (s + 2 < 64) fetch(s + 2, rA, scA);
    unpackStore(s + 1, rB, scB, lds[(s + 1) & 1]);
    compute(s, lds[s & 1]);
    __syncthreads();
    // odd phase: rA holds raw(s+2)
    if (s + 3 < 64) fetch(s + 3, rB, scB);
    if (s + 2 < 64) unpackStore(s + 2, rA, scA, lds[s & 1]);
    compute(s + 1, lds[(s + 1) & 1]);
    __syncthreads();
  }

  // ---- epilogue: C/D layout — VGPR r: lanes 0-15 -> M=r, lanes 16-31 -> M=r+8
  const int ncol = n0 + nsub * 16 + (lane & 15);
  const int mhi  = (lane >> 4) ? 8 : 0;
  const float bv = bias[ncol];
  #pragma unroll
  for (int t2 = 0; t2 < 2; ++t2) {
    const int mt = mhalf * 2 + t2;
    #pragma unroll
    for (int r = 0; r < 8; ++r) {
      out[(size_t)(mt * 16 + r + mhi) * OUT_F + ncol] = acc[t2][r] + bv;
    }
  }
}

extern "C" void kernel_launch(void* const* d_in, const int* in_sizes, int n_in,
                              void* d_out, int out_size, void* d_ws, size_t ws_size,
                              hipStream_t stream) {
  const float* x    = (const float*)d_in[0];
  const int*   Wh   = (const int*)  d_in[1];
  const int*   Wl   = (const int*)  d_in[2];
  const float* sh   = (const float*)d_in[3];
  const float* slo  = (const float*)d_in[4];
  const int*   ci   = (const int*)  d_in[5];
  const float* bias = (const float*)d_in[6];
  float*       out  = (float*)d_out;

  unsigned int* frag = (unsigned int*)d_ws;   // 128*4*32*8 dwords = 512 KB

  // 131072 A-fragment dwords, one per thread
  cpr_build_a_frags<<<512, 256, 0, stream>>>(x, ci, frag);

  // 11008 / 32 = 344 workgroups, 4 waves each, 1376 waves total
  cpr_gemm<<<344, 128, 0, stream>>>(frag, Wh, Wl, sh, slo, bias, out);
}
